// LinearAttention_67903432949803
// MI455X (gfx1250) — matrix operand order repaired and checked by reading, not compiled, not run on hardware
//
#include <hip/hip_runtime.h>

typedef _Float16 half_t;
typedef __attribute__((ext_vector_type(16))) _Float16 v16h;
typedef __attribute__((ext_vector_type(8)))  _Float16 v8h;
typedef __attribute__((ext_vector_type(8)))  float    v8f;

#define WMMA_F16(a, b, c) \
  __builtin_amdgcn_wmma_f32_16x16x32_f16(false, (a), false, (b), (short)0, (c), false, false)

constexpr int kB    = 32;    // batch
constexpr int kC    = 128;   // channels
constexpr int kH    = 4;     // heads
constexpr int kDH   = 32;    // dim per head
constexpr int kHid  = 128;   // heads*dh
constexpr int kN    = 4096;  // tokens (H*W)
constexpr int kTile = 128;   // tokens per LDS tile
constexpr int kGrp  = 8;     // token groups for kernel 1 (512 tokens each)
constexpr int LDX   = 136;   // padded LDS stride (halves): 272 B rows, 16B aligned

// ---- WMMA fragment loaders (CDNA5 ISA 7.12.2 layouts, wave32) ----
// A: 16x32 f16 from row-major [rows][ld] half source (LDS or global).
// Lane m holds K = k0 + kh*8 + {0..7, 16..23}  -> two contiguous 16B chunks.
__device__ __forceinline__ v16h load_frag_a(const half_t* base, int ld,
                                            int row0, int k0, int lane) {
  const int m  = lane & 15;
  const int kh = (lane >> 4) & 1;
  const half_t* p = base + (row0 + m) * ld + k0 + kh * 8;
  v16h f;
#pragma unroll
  for (int j = 0; j < 16; ++j)
    f[j] = p[(j & 7) + ((j >> 3) << 4)];
  return f;
}
// A: 16x32 from row-major FP32 [rows][ld] global source, convert to f16 in flight.
__device__ __forceinline__ v16h load_frag_a_g32(const float* base, int ld,
                                                int row0, int k0, int lane) {
  const int m  = lane & 15;
  const int kh = (lane >> 4) & 1;
  const float* p = base + (row0 + m) * ld + k0 + kh * 8;
  v16h f;
#pragma unroll
  for (int j = 0; j < 16; ++j)
    f[j] = (half_t)p[(j & 7) + ((j >> 3) << 4)];
  return f;
}
// B: 32x16 f16 where source is stored TRANSPOSED: [N][ld] rows, element B[k][n] at
// base[n*ld + k]. Lane n reads K = k0 + kh*8 + {0..7, 16..23} -> two contiguous 16B chunks.
__device__ __forceinline__ v16h load_frag_bT(const half_t* base, int ld,
                                             int k0, int col0, int lane) {
  const int n  = lane & 15;
  const int kh = (lane >> 4) & 1;
  const half_t* p = base + (col0 + n) * ld + k0 + kh * 8;
  v16h f;
#pragma unroll
  for (int j = 0; j < 16; ++j)
    f[j] = p[(j & 7) + ((j >> 3) << 4)];
  return f;
}

// ================= Kernel 1: RMSNorm + K/V proj + exp(k) + partial context =================
// grid (b, h, group), block 256 (8 waves), 4 tiles of 128 tokens per block.
__global__ __launch_bounds__(256)
void k1_context(const float* __restrict__ x, const float* __restrict__ g,
                const float* __restrict__ wqkv,
                float* __restrict__ ctx_part, float* __restrict__ rowsum_part) {
  __shared__ half_t sXt[kTile * LDX];   // xn token-major [t][c]; later ek rows 0-31, ve rows 32-63
  __shared__ float  sNorm[384];

  const int b = blockIdx.x, h = blockIdx.y, grp = blockIdx.z;
  const int tid = threadIdx.x, lane = tid & 31, wave = tid >> 5;
  const float* xb = x + (size_t)b * kC * kN;
  const float  cs = sqrtf((float)kC);

  // GEMM1 mapping: wave -> mt = wave>>1 (0..3: rows of [k(32); v(32)]), nt = (wave&1)*4 + j
  const int mt1 = wave >> 1;
  const float* wbase = (mt1 < 2)
      ? (wqkv + (size_t)(kHid + h * kDH) * kC)            // W_k rows (16 rows at mt1*16)
      : (wqkv + (size_t)(2 * kHid + h * kDH - 32) * kC);  // W_v rows (row index mt1*16-32 rebased)

  v8f ctx = (v8f)0.0f;   // waves 0-3: one 16x16 tile of the 32x32 context
  float rsum = 0.0f;     // tid<32

  for (int it = 0; it < kN / (kGrp * kTile); ++it) {
    const int n0 = grp * 512 + it * kTile;
    __syncthreads();   // previous iteration's sXt readers done

    { // per-token sum of squares (2 threads per token)
      int t = tid & 127, hf = tid >> 7;
      float s = 0.0f;
      for (int c = hf * 64; c < hf * 64 + 64; ++c) {
        float v = xb[(size_t)c * kN + n0 + t];
        s += v * v;
      }
      sNorm[128 + tid] = s;
    }
    __syncthreads();
    if (tid < 128)
      sNorm[tid] = cs * rsqrtf(fmaxf(sNorm[128 + tid] + sNorm[256 + tid], 1e-24f));
    __syncthreads();
    // xn token-major: coalesced global read over t, transposed LDS write
    for (int i = tid; i < kC * kTile; i += 256) {
      int c = i >> 7, t = i & 127;
      sXt[t * LDX + c] = (half_t)(xb[(size_t)c * kN + n0 + t] * sNorm[t] * g[c]);
    }
    __syncthreads();

    // GEMM1: [64 x 128tok] = W_kv * xn. A from global (L2-resident), B from LDS.
    v8f acc[4];
#pragma unroll
    for (int j = 0; j < 4; ++j) acc[j] = (v8f)0.0f;
#pragma unroll
    for (int ks = 0; ks < 4; ++ks) {
      v16h a = load_frag_a_g32(wbase, kC, mt1 * 16, ks * 32, lane);
#pragma unroll
      for (int j = 0; j < 4; ++j) {
        int nt = (wave & 1) * 4 + j;
        v16h bf = load_frag_bT(sXt, LDX, ks * 32, nt * 16, lane);
        acc[j] = WMMA_F16(a, bf, acc[j]);
      }
    }
    __syncthreads();   // done reading sXt

    // ek = exp(k): rows [d][t]; ve: rows [e][t] (both row-major over t, stride LDX)
    half_t* ek = sXt;
    half_t* ve = sXt + 32 * LDX;
    {
      int hi = lane >> 4, nl = lane & 15;
#pragma unroll
      for (int j = 0; j < 4; ++j) {
        int nt = (wave & 1) * 4 + j;
#pragma unroll
        for (int r = 0; r < 8; ++r) {
          int m = mt1 * 16 + hi * 8 + r;   // 0..63
          int t = nt * 16 + nl;
          float v = acc[j][r];
          if (m < 32) ek[m * LDX + t] = (half_t)__expf(v);
          else        ve[(m - 32) * LDX + t] = (half_t)v;
        }
      }
    }
    __syncthreads();

    // rowsum of ek (vectorized 8-half loads)
    if (tid < kDH) {
      const v8h* row = (const v8h*)(ek + tid * LDX);
      float s = 0.0f;
#pragma unroll
      for (int t8 = 0; t8 < kTile / 8; ++t8) {
        v8h v = row[t8];
#pragma unroll
        for (int e = 0; e < 8; ++e) s += (float)v[e];
      }
      rsum += s;
    }
    // GEMM2: ctx[32x32] += ek[32 x t] * v[t x 32]; A=ek rows, B=ve (stored [e][t])
    if (wave < 4) {
      int mt2 = wave >> 1, nt2 = wave & 1;
#pragma unroll
      for (int ks = 0; ks < 4; ++ks) {
        v16h a  = load_frag_a(ek, LDX, mt2 * 16, ks * 32, lane);
        v16h bf = load_frag_bT(ve, LDX, ks * 32, nt2 * 16, lane);
        ctx = WMMA_F16(a, bf, ctx);
      }
    }
  }
  __syncthreads();

  // deterministic partial outputs (unique slots per (b,h,grp))
  if (wave < 4) {
    int mt2 = wave >> 1, nt2 = wave & 1;
    int hi = lane >> 4, nl = lane & 15;
    float* dst = ctx_part + ((size_t)((b * kH + h) * kGrp + grp)) * kDH * kDH;
#pragma unroll
    for (int r = 0; r < 8; ++r) {
      int d = mt2 * 16 + hi * 8 + r, e = nt2 * 16 + nl;
      dst[d * kDH + e] = ctx[r];
    }
  }
  if (tid < kDH)
    rowsum_part[((b * kH + h) * kGrp + grp) * kDH + tid] = rsum;
}

// ================= Kernel 2: reduce partials + memory-kv slots + normalize =================
__global__ __launch_bounds__(256)
void k2_finalize(const float* __restrict__ ctx_part, const float* __restrict__ rowsum_part,
                 const float* __restrict__ mem_kv, half_t* __restrict__ ctxT) {
  const int b = blockIdx.x, h = blockIdx.y, tid = threadIdx.x;
  __shared__ float emk[kDH * 4];
  __shared__ float denom[kDH];
  if (tid < kDH * 4) {
    int d = tid >> 2, s = tid & 3;
    emk[tid] = __expf(mem_kv[((0 * kH + h) * kDH + d) * 4 + s]);
  }
  __syncthreads();
  if (tid < kDH) {
    float s = emk[tid * 4] + emk[tid * 4 + 1] + emk[tid * 4 + 2] + emk[tid * 4 + 3];
    const float* rp = rowsum_part + (size_t)(b * kH + h) * kGrp * kDH + tid;
    for (int gi = 0; gi < kGrp; ++gi) s += rp[gi * kDH];
    denom[tid] = s;
  }
  __syncthreads();
  for (int i = tid; i < kDH * kDH; i += 256) {
    int d = i >> 5, e = i & 31;
    float s = 0.0f;
    const float* cp = ctx_part + (size_t)(b * kH + h) * kGrp * kDH * kDH + i;
    for (int gi = 0; gi < kGrp; ++gi) s += cp[gi * kDH * kDH];
    for (int sl = 0; sl < 4; ++sl)
      s += emk[d * 4 + sl] * mem_kv[((1 * kH + h) * kDH + e) * 4 + sl];
    // store transposed [e][d] as f16 (WMMA A-matrix for kernel 3)
    ctxT[(((size_t)b * kH + h) * kDH + e) * kDH + d] = (half_t)(s / denom[d]);
  }
}

// ================= Kernel 3: q-proj + softmax_d + ctx^T*q + output proj =================
// grid (b, token tile), block 256 (8 waves). mt = wave for all GEMMs.
__global__ __launch_bounds__(256)
void k3_output(const float* __restrict__ x, const float* __restrict__ g,
               const float* __restrict__ wqkv, const half_t* __restrict__ ctxT,
               const float* __restrict__ wout, const float* __restrict__ bout,
               float* __restrict__ out) {
  __shared__ half_t sXt[kTile * LDX];   // token-major [t][c]: xn -> exp(q) -> OH
  __shared__ float  sNorm[384];

  const int b = blockIdx.x;
  const int n0 = blockIdx.y * kTile;
  const int tid = threadIdx.x, lane = tid & 31, wave = tid >> 5;
  const float* xb = x + (size_t)b * kC * kN;

  { // RMSNorm scales
    int t = tid & 127, hf = tid >> 7;
    float s = 0.0f;
    for (int c = hf * 64; c < hf * 64 + 64; ++c) {
      float v = xb[(size_t)c * kN + n0 + t];
      s += v * v;
    }
    sNorm[128 + tid] = s;
  }
  __syncthreads();
  if (tid < 128)
    sNorm[tid] = sqrtf((float)kC) * rsqrtf(fmaxf(sNorm[128 + tid] + sNorm[256 + tid], 1e-24f));
  __syncthreads();
  for (int i = tid; i < kC * kTile; i += 256) {
    int c = i >> 7, t = i & 127;
    sXt[t * LDX + c] = (half_t)(xb[(size_t)c * kN + n0 + t] * sNorm[t] * g[c]);
  }
  __syncthreads();

  // GEMM Q: [128 x 128tok] = Wq * xn. A from global wqkv rows 0..127.
  v8f acc[8];
#pragma unroll
  for (int j = 0; j < 8; ++j) acc[j] = (v8f)0.0f;
#pragma unroll
  for (int ks = 0; ks < 4; ++ks) {
    v16h a = load_frag_a_g32(wqkv, kC, wave * 16, ks * 32, lane);
#pragma unroll
    for (int j = 0; j < 8; ++j) {
      v16h bf = load_frag_bT(sXt, LDX, ks * 32, j * 16, lane);
      acc[j] = WMMA_F16(a, bf, acc[j]);
    }
  }
  __syncthreads();
  // exp(q) written token-major: element (m = wave*16+hi*8+r, t) -> sXt[t][m]
  {
    int hi = lane >> 4, nl = lane & 15;
#pragma unroll
    for (int j = 0; j < 8; ++j)
#pragma unroll
      for (int r = 0; r < 8; ++r)
        sXt[(j * 16 + nl) * LDX + wave * 16 + hi * 8 + r] = (half_t)__expf(acc[j][r]);
  }
  __syncthreads();
  // softmax over d within each head (contiguous 32 halves per (t,h)), * 1/sqrt(dh)
  for (int i = tid; i < kTile * kH; i += 256) {
    int t = i & 127, hh = i >> 7;
    half_t* base = sXt + t * LDX + hh * kDH;
    float s = 0.0f;
#pragma unroll
    for (int d = 0; d < kDH; ++d) s += (float)base[d];
    float r = 0.1767766952966369f / s;   // (1/sqrt(32)) / sum
#pragma unroll
    for (int d = 0; d < kDH; ++d) base[d] = (half_t)((float)base[d] * r);
  }
  __syncthreads();

  // GEMM OH: per head, OH[e][t] = ctx^T[e][d] * q[d][t]; K=32, A from global ctxT (f16)
  const int hh = wave >> 1;
  const half_t* cbase = ctxT + (size_t)(b * kH + hh) * kDH * kDH;
  v8f acc2[8];
#pragma unroll
  for (int j = 0; j < 8; ++j) acc2[j] = (v8f)0.0f;
  {
    v16h a = load_frag_a(cbase, kDH, (wave & 1) * 16, 0, lane);
#pragma unroll
    for (int j = 0; j < 8; ++j) {
      v16h bf = load_frag_bT(sXt + hh * kDH, LDX, 0, j * 16, lane);
      acc2[j] = WMMA_F16(a, bf, acc2[j]);
    }
  }
  __syncthreads();   // all waves done reading q
  // OH token-major over sXt
  {
    int hi = lane >> 4, nl = lane & 15;
#pragma unroll
    for (int j = 0; j < 8; ++j)
#pragma unroll
      for (int r = 0; r < 8; ++r)
        sXt[(j * 16 + nl) * LDX + wave * 16 + hi * 8 + r] = (half_t)acc2[j][r];
  }
  __syncthreads();

  // GEMM OUT: out = Wout * OH + bias. A from global wout.
#pragma unroll
  for (int j = 0; j < 8; ++j) acc[j] = (v8f)0.0f;
#pragma unroll
  for (int ks = 0; ks < 4; ++ks) {
    v16h a = load_frag_a_g32(wout, kHid, wave * 16, ks * 32, lane);
#pragma unroll
    for (int j = 0; j < 8; ++j) {
      v16h bf = load_frag_bT(sXt, LDX, ks * 32, j * 16, lane);
      acc[j] = WMMA_F16(a, bf, acc[j]);
    }
  }
  {
    int hi = lane >> 4, nl = lane & 15;
#pragma unroll
    for (int j = 0; j < 8; ++j)
#pragma unroll
      for (int r = 0; r < 8; ++r) {
        int m = wave * 16 + hi * 8 + r, t = j * 16 + nl;
        out[((size_t)b * kC + m) * kN + n0 + t] = acc[j][r] + bout[m];
      }
  }
}

extern "C" void kernel_launch(void* const* d_in, const int* in_sizes, int n_in,
                              void* d_out, int out_size, void* d_ws, size_t ws_size,
                              hipStream_t stream) {
  const float* x     = (const float*)d_in[0];
  const float* g     = (const float*)d_in[1];
  const float* wqkv  = (const float*)d_in[2];
  const float* memkv = (const float*)d_in[3];
  const float* wout  = (const float*)d_in[4];
  const float* bout  = (const float*)d_in[5];
  float* out = (float*)d_out;

  char* ws = (char*)d_ws;
  float*  ctx_part    = (float*)ws;                         // 32*4*8*32*32*4  = 4 MiB
  float*  rowsum_part = (float*)(ws + 4194304);             // 32*4*8*32*4     = 128 KiB
  half_t* ctxT        = (half_t*)(ws + 4194304 + 131072);   // 32*4*32*32*2    = 256 KiB

  k1_context<<<dim3(kB, kH, kGrp), 256, 0, stream>>>(x, g, wqkv, ctx_part, rowsum_part);
  k2_finalize<<<dim3(kB, kH), 256, 0, stream>>>(ctx_part, rowsum_part, memkv, ctxT);
  k3_output<<<dim3(kB, kN / kTile), 256, 0, stream>>>(x, g, wqkv, ctxT, wout, bout, out);
}
